// ForTorPredictorNN_23175643529937
// MI455X (gfx1250) — matrix operand order repaired and checked by reading, not compiled, not run on hardware
//
#include <hip/hip_runtime.h>
#include <hip/hip_bf16.h>

typedef _Float16 v16h __attribute__((ext_vector_type(16)));
typedef _Float16 v8h  __attribute__((ext_vector_type(8)));
typedef float    v8f  __attribute__((ext_vector_type(8)));

#define N_PART   2048
#define NN       32
#define BATCH    4
#define HDIM     256
#define M_TOTAL  (BATCH * N_PART * NN)   // 262144 rows
#define M_TILE   128                     // rows per block (4 particles)
#define FK       96                      // padded feature K (77 -> 96, mult of 32)
#define FSTR     104                     // feature row stride in halfs
#define ASTR     264                     // activation row stride in halfs (256 + 8 pad)

// ---------------- workspace layout (bytes) ----------------
#define WS_W0T   0                        // 256*96  f16 = 49152
#define WS_WH0T  49152                    // 256*256 f16 = 131072
#define WS_WH1T  180224                   // 256*256 f16 = 131072
#define WS_B0    311296                   // 256 f32
#define WS_BH0   312320                   // 256 f32
#define WS_BH1   313344                   // 256 f32  (total 314368 B)

// ---------------------------------------------------------------------------
// Prep: fold BN (eval mode, running stats (0,1)) into weights/bias, transpose
// to [out_col][k] f16 layout for direct WMMA B-fragment loads.
// ---------------------------------------------------------------------------
__global__ void prep_weights(const float* __restrict__ w0, const float* __restrict__ b0,
                             const float* __restrict__ wh, const float* __restrict__ bh,
                             const float* __restrict__ gamma, const float* __restrict__ beta,
                             _Float16* __restrict__ W0t, _Float16* __restrict__ Wh0t,
                             _Float16* __restrict__ Wh1t,
                             float* __restrict__ b0s, float* __restrict__ bh0s,
                             float* __restrict__ bh1s) {
  const int n = blockIdx.x;     // output channel 0..255
  const int k = threadIdx.x;    // input channel  0..255
  const float inv_std = rsqrtf(1.0f + 1e-5f);
  const float s0 = gamma[n] * inv_std;
  const float s1 = gamma[HDIM + n] * inv_std;
  // wh: (2, 256, 256) row-major; wh[i][k][n]
  Wh0t[n * HDIM + k] = (_Float16)(wh[k * HDIM + n] * s0);
  Wh1t[n * HDIM + k] = (_Float16)(wh[HDIM * HDIM + k * HDIM + n] * s1);
  if (k < FK) {
    W0t[n * FK + k] = (_Float16)((k < 77) ? w0[k * HDIM + n] : 0.0f);
  }
  if (n == 0) {
    const float sk0 = gamma[k] * inv_std;
    const float sk1 = gamma[HDIM + k] * inv_std;
    b0s[k]  = b0[k];
    bh0s[k] = bh[k] * sk0 + beta[k];
    bh1s[k] = bh[HDIM + k] * sk1 + beta[HDIM + k];
  }
}

// ---------------------------------------------------------------------------
// Fragment loads.
// A (16-bit, 16x32): VGPR0-3: lanes0-15 K=0..7, lanes16-31 K=8..15;
//                    VGPR4-7: lanes0-15 K=16..23, lanes16-31 K=24..31.
// B (16-bit, 32x16): same K layout as A (col=lane).
// ---------------------------------------------------------------------------
__device__ __forceinline__ v16h ld_fragA(const _Float16* p) {
  union { v8h h[2]; v16h v; } u;
  u.h[0] = *(const v8h*)(p);        // halfs +0..7
  u.h[1] = *(const v8h*)(p + 16);   // halfs +16..23
  return u.v;
}
__device__ __forceinline__ v16h ld_fragB(const _Float16* p) {
  union { v8h h[2]; v16h v; } u;
  u.h[0] = *(const v8h*)(p);        // halfs +0..7
  u.h[1] = *(const v8h*)(p + 16);   // halfs +16..23
  return u.v;
}

// One fused-MLP layer: out[128x256] = relu(in[128xK] @ W[Kx256] + bias).
// Each wave owns 2 col-tiles x all 8 row-tiles:
//   B fragment: 2 global_load_b128, reused across 8 WMMAs (software-pipelined
//   one k-step ahead so global latency hides under wmma + LDS work).
//   A fragments: ds_load_b128 pairs from LDS (low-latency path).
template <int KSTEPS>
__device__ __forceinline__ void gemm_layer(const _Float16* __restrict__ lin, int sIn,
                                           const _Float16* __restrict__ wt, int Kp,
                                           const float* __restrict__ bias,
                                           _Float16* __restrict__ lout) {
  const int lane = threadIdx.x & 31;
  const int wave = threadIdx.x >> 5;   // 0..7
  const int l15  = lane & 15;
  const int hi   = lane >> 4;          // 0 or 1
#pragma unroll
  for (int g = 0; g < 2; ++g) {
    const int ct  = wave * 2 + g;      // col tile 0..15
    const int col = ct * 16 + l15;
    const _Float16* wcol = wt + col * Kp + hi * 8;
    v8f acc[8] = {};
    v16h Bcur = ld_fragB(wcol);        // prefetch k-step 0
#pragma unroll
    for (int kk = 0; kk < KSTEPS; ++kk) {
      v16h Bnxt = Bcur;
      if (kk + 1 < KSTEPS) Bnxt = ld_fragB(wcol + (kk + 1) * 32);  // prefetch next
#pragma unroll
      for (int mr = 0; mr < 8; ++mr) {
        const v16h A = ld_fragA(lin + (mr * 16 + l15) * sIn + kk * 32 + hi * 8);
        acc[mr] = __builtin_amdgcn_wmma_f32_16x16x32_f16(
            /*neg_a=*/false, A, /*neg_b=*/false, Bcur,
            /*c_mod=*/(short)0, acc[mr], /*reuse_a=*/false, /*reuse_b=*/false);
      }
      Bcur = Bnxt;
    }
    const float bv = bias[col];
#pragma unroll
    for (int mr = 0; mr < 8; ++mr) {
      const int rbase = mr * 16 + hi * 8;   // D: VGPR j -> row j (lanes<16) / 8+j
#pragma unroll
      for (int j = 0; j < 8; ++j) {
        float v = acc[mr][j] + bv;
        v = v > 0.0f ? v : 0.0f;
        lout[(rbase + j) * ASTR + col] = (_Float16)v;
      }
    }
  }
}

// ---------------------------------------------------------------------------
// Fused: features -> 3x WMMA layers -> final 256->3 -> neighbor mean pool.
// ---------------------------------------------------------------------------
__global__ __launch_bounds__(256) void fortor_fused(
    const float* __restrict__ pos, const float* __restrict__ ori,
    const int* __restrict__ nl,
    const _Float16* __restrict__ W0t, const _Float16* __restrict__ Wh0t,
    const _Float16* __restrict__ Wh1t,
    const float* __restrict__ b0s, const float* __restrict__ bh0s,
    const float* __restrict__ bh1s,
    const float* __restrict__ wf, const float* __restrict__ bf,
    float* __restrict__ out) {
  __shared__ _Float16 ldsA[M_TILE * ASTR];   // 67584 B
  __shared__ _Float16 ldsB[M_TILE * ASTR];   // 67584 B (also hosts features)
  const int tid = threadIdx.x;
  const int mbase = blockIdx.x * M_TILE;

  // ---- stage 1: features (77 values + zero pad to 96) into ldsB ----
  if (tid < M_TILE) {
    const int m  = mbase + tid;
    const int b  = m >> 16;            // m / (N*Nn)
    const int pi = m >> 5;             // flat particle index b*N + n
    const int nidx = nl[2 * m + 1];
    const int pj = b * N_PART + nidx;

    float P[3], Q[3], dr[3];
#pragma unroll
    for (int c = 0; c < 3; ++c) { P[c] = pos[pi * 3 + c]; Q[c] = pos[pj * 3 + c]; }
#pragma unroll
    for (int c = 0; c < 3; ++c) {
      float d = Q[c] - P[c];
      d -= 20.0f * rintf(d * 0.05f);   // minimum-image PBC, round-half-even
      dr[c] = d;
    }
    const float R2 = dr[0] * dr[0] + dr[1] * dr[1] + dr[2] * dr[2];
    const float R = sqrtf(R2);
    const float invR = 1.0f / R;

    float Rp[9], Rn[9];                // [k*3 + i] row-major
#pragma unroll
    for (int k = 0; k < 9; ++k) { Rp[k] = ori[pi * 9 + k]; Rn[k] = ori[pj * 9 + k]; }

    _Float16* f = ldsB + tid * FSTR;
    f[0] = (_Float16)R;
#pragma unroll
    for (int c = 0; c < 3; ++c) f[1 + c] = (_Float16)(dr[c] * invR);
    f[4] = (_Float16)invR;
    // dot[i][j] = sum_k Rp[k][i]*Rn[k][j]
#pragma unroll
    for (int i = 0; i < 3; ++i)
#pragma unroll
      for (int j = 0; j < 3; ++j) {
        float s = 0.0f;
#pragma unroll
        for (int k = 0; k < 3; ++k) s += Rp[k * 3 + i] * Rn[k * 3 + j];
        f[5 + i * 3 + j] = (_Float16)s;
      }
    // ep[i][j][c] = Rp[c][i]*Rn[c][j];  ep_norm[i][j]
#pragma unroll
    for (int i = 0; i < 3; ++i)
#pragma unroll
      for (int j = 0; j < 3; ++j) {
        float ss = 0.0f;
#pragma unroll
        for (int c = 0; c < 3; ++c) {
          const float e = Rp[c * 3 + i] * Rn[c * 3 + j];
          f[14 + (i * 3 + j) * 3 + c] = (_Float16)e;
          ss += e * e;
        }
        f[41 + i * 3 + j] = (_Float16)sqrtf(ss);
      }
    // cp[i] = cross(colP_i, colN_i); cp_norm[i]
#pragma unroll
    for (int i = 0; i < 3; ++i) {
      const float a0 = Rp[0 * 3 + i], a1 = Rp[1 * 3 + i], a2 = Rp[2 * 3 + i];
      const float b0v = Rn[0 * 3 + i], b1 = Rn[1 * 3 + i], b2 = Rn[2 * 3 + i];
      const float c0 = a1 * b2 - a2 * b1;
      const float c1 = a2 * b0v - a0 * b2;
      const float c2 = a0 * b1 - a1 * b0v;
      f[50 + i * 3 + 0] = (_Float16)c0;
      f[50 + i * 3 + 1] = (_Float16)c1;
      f[50 + i * 3 + 2] = (_Float16)c2;
      f[59 + i] = (_Float16)sqrtf(c0 * c0 + c1 * c1 + c2 * c2);
    }
    // rel[i][j] = sum_k Rp[i][k]*Rn[j][k]
#pragma unroll
    for (int i = 0; i < 3; ++i)
#pragma unroll
      for (int j = 0; j < 3; ++j) {
        float s = 0.0f;
#pragma unroll
        for (int k = 0; k < 3; ++k) s += Rp[i * 3 + k] * Rn[j * 3 + k];
        f[62 + i * 3 + j] = (_Float16)s;
      }
    // rbf_p[k] = exp(-(dr . colP_k)^2); rbf_n likewise
#pragma unroll
    for (int k = 0; k < 3; ++k) {
      const float pp = dr[0] * Rp[0 * 3 + k] + dr[1] * Rp[1 * 3 + k] + dr[2] * Rp[2 * 3 + k];
      const float pn = dr[0] * Rn[0 * 3 + k] + dr[1] * Rn[1 * 3 + k] + dr[2] * Rn[2 * 3 + k];
      f[71 + k] = (_Float16)expf(-pp * pp);
      f[74 + k] = (_Float16)expf(-pn * pn);
    }
#pragma unroll
    for (int k = 77; k < FK; ++k) f[k] = (_Float16)0.0f;
  }
  __syncthreads();

  // ---- stage 2: three WMMA GEMM layers, LDS ping-pong ----
  gemm_layer<3>(ldsB, FSTR, W0t, FK, b0s, ldsA);
  __syncthreads();
  gemm_layer<8>(ldsA, ASTR, Wh0t, HDIM, bh0s, ldsB);
  __syncthreads();
  gemm_layer<8>(ldsB, ASTR, Wh1t, HDIM, bh1s, ldsA);
  __syncthreads();

  // ---- stage 3: final 256->3 + neighbor-mean pooling ----
  // Waves 0..3: each wave == one particle's 32 neighbor rows.
  if (tid < M_TILE) {
    const _Float16* h = ldsA + tid * ASTR;
    float a0 = 0.0f, a1 = 0.0f, a2 = 0.0f;
    for (int k = 0; k < HDIM; ++k) {
      const float x = (float)h[k];
      a0 += x * wf[3 * k + 0];
      a1 += x * wf[3 * k + 1];
      a2 += x * wf[3 * k + 2];
    }
#pragma unroll
    for (int off = 16; off > 0; off >>= 1) {
      a0 += __shfl_down(a0, off, 32);
      a1 += __shfl_down(a1, off, 32);
      a2 += __shfl_down(a2, off, 32);
    }
    if ((tid & 31) == 0) {
      const int p = (mbase >> 5) + (tid >> 5);   // flat particle index
      out[p * 3 + 0] = a0 * (1.0f / NN) + bf[0];
      out[p * 3 + 1] = a1 * (1.0f / NN) + bf[1];
      out[p * 3 + 2] = a2 * (1.0f / NN) + bf[2];
    }
  }
}

extern "C" void kernel_launch(void* const* d_in, const int* in_sizes, int n_in,
                              void* d_out, int out_size, void* d_ws, size_t ws_size,
                              hipStream_t stream) {
  const float* position = (const float*)d_in[0];
  const float* orient   = (const float*)d_in[1];
  const int*   nlist    = (const int*)d_in[2];
  const float* w0    = (const float*)d_in[3];
  const float* b0    = (const float*)d_in[4];
  const float* wh    = (const float*)d_in[5];
  const float* bh    = (const float*)d_in[6];
  const float* gamma = (const float*)d_in[7];
  const float* beta  = (const float*)d_in[8];
  const float* wf    = (const float*)d_in[9];
  const float* bf    = (const float*)d_in[10];
  float* out = (float*)d_out;

  char* ws = (char*)d_ws;
  _Float16* W0t  = (_Float16*)(ws + WS_W0T);
  _Float16* Wh0t = (_Float16*)(ws + WS_WH0T);
  _Float16* Wh1t = (_Float16*)(ws + WS_WH1T);
  float* b0s  = (float*)(ws + WS_B0);
  float* bh0s = (float*)(ws + WS_BH0);
  float* bh1s = (float*)(ws + WS_BH1);

  prep_weights<<<dim3(HDIM), dim3(HDIM), 0, stream>>>(
      w0, b0, wh, bh, gamma, beta, W0t, Wh0t, Wh1t, b0s, bh0s, bh1s);

  fortor_fused<<<dim3(M_TOTAL / M_TILE), dim3(256), 0, stream>>>(
      position, orient, nlist, W0t, Wh0t, Wh1t, b0s, bh0s, bh1s, wf, bf, out);
}
